// winconv2d_fi_30889404793386
// MI455X (gfx1250) — matrix operand order repaired and checked by reading, not compiled, not run on hardware
//
#include <hip/hip_runtime.h>

typedef __attribute__((ext_vector_type(16))) _Float16 v16h_t;
typedef __attribute__((ext_vector_type(8)))  float    v8f_t;

#define CI_   32
#define CO_   32
#define HH_   56
#define NC_   28                 // output tiles per side
#define NT_   (NC_*NC_)          // 784 tiles per image
#define NB_   4                  // batch
#define MTOT_ (NB_*NT_)          // 3136 total M rows
#define MBLK_ (MTOT_/16)         // 196 blocks of 16 rows (exact)

// ---------------------------------------------------------------------------
// Kernel 1: quantize weights (global max-abs, RNE round), Winograd-transform
// Wt = G w G^T, and pack straight into the WMMA B-fragment register layout:
//   B value (K=ci, N=oc) at position p -> wsB[((p*2+nb)*32 + lane)*16 + e]
//   nb = oc/16, lane = (oc%16) + 16*((ci%16)/8), e = ci%8 + 8*(ci/16)  (ISA 7.12.2)
// ---------------------------------------------------------------------------
__global__ void winconv_wt_kernel(const float* __restrict__ w,
                                  _Float16* __restrict__ wsB) {
  __shared__ float red[256];
  __shared__ float s_scale;
  const int tid = threadIdx.x;

  float mx = 0.f;
  for (int i = tid; i < CO_*CI_*9; i += 256) mx = fmaxf(mx, fabsf(w[i]));
  red[tid] = mx;
  __syncthreads();
  for (int s = 128; s > 0; s >>= 1) {
    if (tid < s) red[tid] = fmaxf(red[tid], red[tid + s]);
    __syncthreads();
  }
  if (tid == 0) s_scale = red[0] * (1.0f / 127.0f);
  __syncthreads();

  const float scale = s_scale;
  const float safe  = (scale > 0.f) ? scale : 1.0f;

  for (int pair = tid; pair < CO_*CI_; pair += 256) {
    const int oc = pair >> 5, ci = pair & 31;
    float g[3][3];
#pragma unroll
    for (int i = 0; i < 3; ++i)
#pragma unroll
      for (int j = 0; j < 3; ++j) {
        const float v = w[pair*9 + i*3 + j];
        const float q = rintf(v / safe) * safe;   // RNE, matches jnp.round
        g[i][j] = (scale > 0.f) ? q : v;
      }
    // t = G * g (4x3)
    float t[4][3];
#pragma unroll
    for (int j = 0; j < 3; ++j) {
      t[0][j] = g[0][j];
      t[1][j] = 0.5f*(g[0][j] + g[1][j] + g[2][j]);
      t[2][j] = 0.5f*(g[0][j] - g[1][j] + g[2][j]);
      t[3][j] = g[2][j];
    }
    const int nb   = oc >> 4;
    const int lane = (oc & 15) + (((ci & 15) >> 3) << 4);
    const int e    = (ci & 7) + ((ci >> 4) << 3);
#pragma unroll
    for (int i = 0; i < 4; ++i) {
      float wt[4];   // row i of  t * G^T
      wt[0] = t[i][0];
      wt[1] = 0.5f*(t[i][0] + t[i][1] + t[i][2]);
      wt[2] = 0.5f*(t[i][0] - t[i][1] + t[i][2]);
      wt[3] = t[i][2];
#pragma unroll
      for (int l = 0; l < 4; ++l) {
        const int p = i*4 + l;
        wsB[((p*2 + nb)*32 + lane)*16 + e] = (_Float16)wt[l];
      }
    }
  }
}

// ---------------------------------------------------------------------------
// Kernel 2: one wave32 per block handles 16 Winograd tiles (one 16x16x32 WMMA
// A-tile) for all 32 output channels.  Phase 1 builds the 16 A fragments in
// LDS in exact A-fragment layout (M=tile: lane = M + 16*((ci%16)/8),
// e = ci%8 + 8*(ci/16)); phase 2 runs 16p x 2nb = 32 WMMAs and folds the
// A^T y A output transform into {-1,0,1}-coefficient accumulation.
// The p-loop is kept rolled (#pragma unroll 1) so each C fragment is folded
// into the 2x2 accumulators before the next WMMA issues -> no VGPR spills.
// ---------------------------------------------------------------------------
__global__ void __launch_bounds__(32)
winconv_main_kernel(const float* __restrict__ x,
                    const _Float16* __restrict__ wsB,
                    float* __restrict__ out) {
  extern __shared__ _Float16 ldsA[];   // [16 p][32 lane][16 e] = 16 KB
  const int lane   = threadIdx.x;      // 0..31
  const int mblock = blockIdx.x;       // 0..195

  // ---- phase 1: input tile transform Bt X B -> LDS (A fragments) ----
  for (int it = 0; it < 16; ++it) {
    const int tile = lane & 15;                 // 16 tiles
    const int ci   = (it << 1) | (lane >> 4);   // 2 channels per iteration
    const int m    = mblock*16 + tile;
    const int b    = m / NT_;
    const int t    = m % NT_;
    const int tr = t / NC_, tc = t % NC_;
    const int r0 = tr*2 - 1, c0 = tc*2 - 1;     // padded origin
    const float* xp = x + (b*CI_ + ci)*(HH_*HH_);

    float X[4][4];
#pragma unroll
    for (int di = 0; di < 4; ++di) {
      const int row = r0 + di;
#pragma unroll
      for (int dj = 0; dj < 4; ++dj) {
        const int col = c0 + dj;
        const bool ok = (row >= 0) & (row < HH_) & (col >= 0) & (col < HH_);
        X[di][dj] = ok ? xp[row*HH_ + col] : 0.f;
      }
    }
    float t1[4][4];                              // Bt * X
#pragma unroll
    for (int j = 0; j < 4; ++j) {
      t1[0][j] = X[0][j] - X[2][j];
      t1[1][j] = X[1][j] + X[2][j];
      t1[2][j] = X[2][j] - X[1][j];
      t1[3][j] = X[1][j] - X[3][j];
    }
    const int half  = (ci & 15) >> 3;
    const int lanep = tile + (half << 4);
    const int e     = (ci & 7) + ((ci >> 4) << 3);
#pragma unroll
    for (int i = 0; i < 4; ++i) {                // (Bt X) * B, scatter to LDS
      float xt[4];
      xt[0] = t1[i][0] - t1[i][2];
      xt[1] = t1[i][1] + t1[i][2];
      xt[2] = t1[i][2] - t1[i][1];
      xt[3] = t1[i][1] - t1[i][3];
#pragma unroll
      for (int l = 0; l < 4; ++l)
        ldsA[((i*4 + l)*32 + lanep)*16 + e] = (_Float16)xt[l];
    }
  }
  // same-wave DS ops are in-order; fence before fragment reads (CDNA5 counter)
  asm volatile("s_wait_dscnt 0" ::: "memory");

  // ---- phase 2: 32x v_wmma_f32_16x16x32_f16 + fused output transform ----
  float acc[2][8][2][2];
#pragma unroll
  for (int nb = 0; nb < 2; ++nb)
#pragma unroll
    for (int r = 0; r < 8; ++r)
#pragma unroll
      for (int u = 0; u < 2; ++u)
#pragma unroll
        for (int v = 0; v < 2; ++v) acc[nb][r][u][v] = 0.f;

#pragma unroll 1          // rolled: fold each C before the next WMMA issues
  for (int p = 0; p < 16; ++p) {
    const v16h_t a = *(const v16h_t*)(ldsA + (p*32 + lane)*16);
    const int i = p >> 2, l = p & 3;
    // AT row0 = [1,1,1,0], row1 = [0,1,-1,1]  (computed, not memory-indexed)
    const float a0i = (i == 3) ? 0.f : 1.f;
    const float a1i = (i == 0) ? 0.f : ((i == 2) ? -1.f : 1.f);
    const float a0l = (l == 3) ? 0.f : 1.f;
    const float a1l = (l == 0) ? 0.f : ((l == 2) ? -1.f : 1.f);
    const float c00 = a0i * a0l, c01 = a0i * a1l;
    const float c10 = a1i * a0l, c11 = a1i * a1l;

#pragma unroll
    for (int nb = 0; nb < 2; ++nb) {
      const v16h_t bf = *(const v16h_t*)(wsB + ((p*2 + nb)*32 + lane)*16);
      v8f_t c = {};
      c = __builtin_amdgcn_wmma_f32_16x16x32_f16(
              /*neg_a=*/false, a, /*neg_b=*/false, bf,
              /*c_mod=*/(short)0, c, /*reuse_a=*/false, /*reuse_b=*/false);
#pragma unroll
      for (int r = 0; r < 8; ++r) {
        acc[nb][r][0][0] += c00 * c[r];
        acc[nb][r][0][1] += c01 * c[r];
        acc[nb][r][1][0] += c10 * c[r];
        acc[nb][r][1][1] += c11 * c[r];
      }
    }
  }

  // ---- store 2x2 outputs: lane holds (M = r + 8*(lane/16), N = lane%16) ----
  const int nL = lane & 15, hL = lane >> 4;
#pragma unroll
  for (int nb = 0; nb < 2; ++nb) {
    const int oc = nL + nb*16;
#pragma unroll
    for (int r = 0; r < 8; ++r) {
      const int m  = mblock*16 + r + hL*8;
      const int b  = m / NT_;
      const int t  = m % NT_;
      const int tr = t / NC_, tc = t % NC_;
      float* op = out + (b*CO_ + oc)*(HH_*HH_) + (tr*2)*HH_ + tc*2;
      op[0]       = acc[nb][r][0][0];
      op[1]       = acc[nb][r][0][1];
      op[HH_]     = acc[nb][r][1][0];
      op[HH_ + 1] = acc[nb][r][1][1];
    }
  }
}

extern "C" void kernel_launch(void* const* d_in, const int* in_sizes, int n_in,
                              void* d_out, int out_size, void* d_ws, size_t ws_size,
                              hipStream_t stream) {
  const float* x = (const float*)d_in[0];       // (4,32,56,56) f32
  const float* w = (const float*)d_in[1];       // (32,32,3,3)  f32
  float*     out = (float*)d_out;               // (4,32,56,56) f32
  _Float16*  wsB = (_Float16*)d_ws;             // 16*2*32*16 halves = 32 KB

  winconv_wt_kernel<<<1, 256, 0, stream>>>(w, wsB);

  const size_t ldsBytes = (size_t)16 * 32 * 16 * sizeof(_Float16);  // 16 KB
  winconv_main_kernel<<<MBLK_, 32, ldsBytes, stream>>>(x, wsB, out);
}
